// ImageMaskingNet_35605278884051
// MI455X (gfx1250) — matrix operand order repaired and checked by reading, not compiled, not run on hardware
//
#include <hip/hip_runtime.h>
#include <hip/hip_bf16.h>

// MI455X fused SNN kernel:
//  - per-(b,h) workgroup: 128x512 GEMM via v_wmma_f32_16x16x32_f16 (fp16 hi/lo
//    split x3 for fp32-grade accuracy), Y staged in LDS (258 KiB), fused
//    bidirectional LIF scan from LDS, single HBM write of the output.
//  - W pre-split into WMMA B-fragment layout in d_ws (1 MiB, L2 resident).

typedef _Float16 v16h __attribute__((ext_vector_type(16)));
typedef _Float16 v2h  __attribute__((ext_vector_type(2)));
typedef float    v8f  __attribute__((ext_vector_type(8)));
typedef float    v4f  __attribute__((ext_vector_type(4)));

#define Hh 128
#define Wn 128          // w dimension (sequence length)
#define Bb 8
#define Dd 512          // K of GEMM
#define Ee 512          // N of GEMM
#define YPITCH 516      // padded LDS row (floats) -> conflict-free C writes
#define XFRAG (8*32*16) // halves per (buf,hi/lo) A-fragment block (8 w-tiles)

// ---------------- W prep: fp32 -> fp16 hi/lo in WMMA B-fragment order -------
// flat idx = (((et*16 + kt)*32 + lane)*16 + i)
// B 32x16 f16 layout:  N = et*16 + (lane&15),  K = kt*32 + (i<8 ? i : i+8) + 8*(lane>=16)
__global__ __launch_bounds__(256) void prep_w_kernel(
    const float* __restrict__ Wlin, _Float16* __restrict__ Whi,
    _Float16* __restrict__ Wlo) {
  int idx  = blockIdx.x * 256 + threadIdx.x;     // 0 .. 512*512-1
  int i    = idx & 15;
  int lane = (idx >> 4) & 31;
  int kt   = (idx >> 9) & 15;
  int et   = idx >> 13;
  int N = et * 16 + (lane & 15);
  int K = kt * 32 + ((i < 8) ? i : i + 8) + ((lane >= 16) ? 8 : 0);
  float w = Wlin[N * Dd + K];
  _Float16 h = (_Float16)w;
  Whi[idx] = h;
  Wlo[idx] = (_Float16)(w - (float)h);
}

// Stage one 128x32 fp32 X chunk into LDS as hi/lo fp16 A-fragments.
// A 16x32 f16 layout: M = lane&15, K = (i<8 ? i : i+8) + 8*(lane>=16)
// inverse: i = (k&7) + (k>=16 ? 8 : 0), laneHi = (k>>3)&1
__device__ __forceinline__ void stage_chunk(_Float16* Xs, int buf,
                                            v4f r0, v4f r1, int lw, int dq) {
  float v[8];
  v[0] = r0[0]; v[1] = r0[1]; v[2] = r0[2]; v[3] = r0[3];
  v[4] = r1[0]; v[5] = r1[1]; v[6] = r1[2]; v[7] = r1[3];
  const int wtT  = lw >> 4;
  const int mrow = lw & 15;
  _Float16* xh = Xs + (buf * 2 + 0) * XFRAG;
  _Float16* xl = Xs + (buf * 2 + 1) * XFRAG;
#pragma unroll
  for (int p = 0; p < 4; ++p) {
    int k0 = dq * 8 + 2 * p;                       // even, pair k0/k0+1
    int i  = (k0 & 7) + ((k0 >= 16) ? 8 : 0);      // even -> dword aligned
    int laneT = mrow | (((k0 >> 3) & 1) << 4);
    float f0 = v[2 * p], f1 = v[2 * p + 1];
    _Float16 h0 = (_Float16)f0, h1 = (_Float16)f1;
    _Float16 l0 = (_Float16)(f0 - (float)h0);
    _Float16 l1 = (_Float16)(f1 - (float)h1);
    int base = wtT * 512 + laneT * 16 + i;
    v2h hp; hp[0] = h0; hp[1] = h1;
    v2h lp; lp[0] = l0; lp[1] = l1;
    *(v2h*)(xh + base) = hp;
    *(v2h*)(xl + base) = lp;
  }
}

// ---------------- fused GEMM (WMMA) + bidirectional LIF scan ----------------
__global__ __launch_bounds__(512, 1) void snn_fused_kernel(
    const float* __restrict__ x, const _Float16* __restrict__ Whi,
    const _Float16* __restrict__ Wlo, const float* __restrict__ blin,
    float* __restrict__ out) {
  extern __shared__ char smem[];
  float*    Ys = (float*)smem;                        // 128 x 516 f32
  _Float16* Xs = (_Float16*)(smem + Wn * YPITCH * 4); // 2 bufs x hi/lo frags

  const int tid  = threadIdx.x;
  const int lane = tid & 31;
  const int wave = tid >> 5;       // 16 waves
  const int bh   = blockIdx.x;     // 1024 workgroups
  const int b    = bh & 7;
  const int h    = bh >> 3;

  const int wt = wave & 7;         // w-tile (M tile) owned by this wave
  const int eg = wave >> 3;        // e-group within pass (0/1)

  // X loader mapping: each thread loads 8 consecutive floats of one w-row
  const int lw = tid >> 2;         // w row (0..127)
  const int dq = tid & 3;          // 8-float group within 32-wide chunk
  const float* xrow =
      x + (((size_t)(h * Wn + lw) * Bb + b) * Dd) + dq * 8;

  for (int ep = 0; ep < 2; ++ep) {  // two passes over e (256 cols each)
    v8f acc[8] = {};                // 8 C tiles of 16x16 per wave

    // prologue: chunk 0 -> buf 0
    v4f r0 = *(const v4f*)(xrow);
    v4f r1 = *(const v4f*)(xrow + 4);
    stage_chunk(Xs, 0, r0, r1, lw, dq);
    __syncthreads();

    for (int kt = 0; kt < 16; ++kt) {
      const int buf = kt & 1;
      if (kt < 15) {                               // prefetch next chunk
        const float* p = xrow + (kt + 1) * 32;
        r0 = *(const v4f*)p;
        r1 = *(const v4f*)(p + 4);
      }
      // A fragments (hi/lo) for this wave's w-tile from LDS
      const _Float16* ap = Xs + (buf * 2) * XFRAG + wt * 512 + lane * 16;
      v16h a_hi = *(const v16h*)ap;
      v16h a_lo = *(const v16h*)(ap + XFRAG);
#pragma unroll
      for (int j = 0; j < 8; ++j) {
        const int et = ep * 16 + eg * 8 + j;
        const int boff = ((et * 16 + kt) * 32 + lane) * 16;
        v16h b_hi = *(const v16h*)(Whi + boff);
        v16h b_lo = *(const v16h*)(Wlo + boff);
        acc[j] = __builtin_amdgcn_wmma_f32_16x16x32_f16(
            false, a_hi, false, b_hi, (short)0, acc[j], false, false);
        acc[j] = __builtin_amdgcn_wmma_f32_16x16x32_f16(
            false, a_hi, false, b_lo, (short)0, acc[j], false, false);
        acc[j] = __builtin_amdgcn_wmma_f32_16x16x32_f16(
            false, a_lo, false, b_hi, (short)0, acc[j], false, false);
      }
      __syncthreads();                             // everyone done with buf
      if (kt < 15) {
        stage_chunk(Xs, buf ^ 1, r0, r1, lw, dq);
        __syncthreads();                           // next buf ready
      }
    }

    // C tiles (+bias) -> LDS Y.  f32 C layout: row = r + 8*laneHi, col = lane&15
    const int laneHi = lane >> 4;
#pragma unroll
    for (int j = 0; j < 8; ++j) {
      const int et  = ep * 16 + eg * 8 + j;
      const int col = et * 16 + (lane & 15);
      const float bias = blin[col];
      const int row0 = wt * 16 + (laneHi << 3);
#pragma unroll
      for (int r = 0; r < 8; ++r)
        Ys[(row0 + r) * YPITCH + col] = acc[j][r] + bias;
    }
    __syncthreads();
  }

  // ---- bidirectional LIF scan straight out of LDS (one thread per e) ----
  {
    const int e = tid;
    float vf = 0.0f, vb = 0.0f;
    unsigned fbs[4], bbs[4];
#pragma unroll
    for (int q = 0; q < 4; ++q) {                  // 4 blocks of 32 steps
      unsigned fm = 0, bm = 0;
      for (int ii = 0; ii < 32; ++ii) {
        const int wf = q * 32 + ii;                // forward position
        const int wb = 127 - wf;                   // backward position
        const float yf = Ys[wf * YPITCH + e];
        const float yb = Ys[wb * YPITCH + e];
        vf = 0.5f * (vf + yf);                     // v += (y - v)/TAU, TAU=2
        if (vf >= 1.0f) { fm |= 1u << ii;        vf = 0.0f; }
        vb = 0.5f * (vb + yb);
        if (vb >= 1.0f) { bm |= 1u << (31 - ii); vb = 0.0f; }
      }
      fbs[q] = fm;
      bbs[3 - q] = bm;                             // wb block = 3-q, bit 31-ii
    }

    float* orow = out + ((size_t)(h * Wn) * Bb + b) * Dd + e;
#pragma unroll
    for (int q = 0; q < 4; ++q)
      for (int ii = 0; ii < 32; ++ii) {
        const int w = q * 32 + ii;
        const float s = (float)((fbs[q] >> ii) & 1u) +
                        (float)((bbs[q] >> ii) & 1u);
        orow[(size_t)w * (Bb * Dd)] = s;           // 128B-contig per wave
      }
  }
}

extern "C" void kernel_launch(void* const* d_in, const int* in_sizes, int n_in,
                              void* d_out, int out_size, void* d_ws,
                              size_t ws_size, hipStream_t stream) {
  const float* x    = (const float*)d_in[0];
  const float* Wlin = (const float*)d_in[1];
  const float* blin = (const float*)d_in[2];
  float* out = (float*)d_out;

  _Float16* Whi = (_Float16*)d_ws;       // 512 KiB
  _Float16* Wlo = Whi + Dd * Ee;         // 512 KiB

  prep_w_kernel<<<(Dd * Ee) / 256, 256, 0, stream>>>(Wlin, Whi, Wlo);

  const size_t lds = (size_t)Wn * YPITCH * 4 + 4 * XFRAG * sizeof(_Float16);
  snn_fused_kernel<<<Hh * Bb, 512, lds, stream>>>(x, Whi, Wlo, blin, out);
}
